// MultiHeadAttention_16192026706444
// MI455X (gfx1250) — matrix operand order repaired and checked by reading, not compiled, not run on hardware
//
#include <hip/hip_runtime.h>
#include <hip/hip_bf16.h>

typedef __bf16 v16bf __attribute__((ext_vector_type(16)));
typedef __bf16 v8bf  __attribute__((ext_vector_type(8)));
typedef float  v8f   __attribute__((ext_vector_type(8)));
typedef unsigned int u32x4 __attribute__((ext_vector_type(4)));
typedef int          i32x4 __attribute__((ext_vector_type(4)));
typedef int          i32x8 __attribute__((ext_vector_type(8)));

union Frag { v16bf v; v8bf h[2]; };

// ---------------------------------------------------------------- WMMA frags
// A-operand fragment (16x32 bf16, MxK): lane row = lane%16,
// k = {b..b+7} U {b+16..b+23}, b = (lane<16)?0:8.
__device__ inline v16bf load_a_frag(const __bf16* row, int lane) {
    const int kb = (lane & 16) ? 8 : 0;
    Frag f;
    f.h[0] = *(const v8bf*)(row + kb);
    f.h[1] = *(const v8bf*)(row + kb + 16);
    return f.v;
}

// B-operand fragment (32x16 bf16, KxN) from a transposed (NT) tile:
// lane column = lane%16, k = {b..b+7} U {b+16..b+23}, b = (lane<16)?0:8.
__device__ inline v16bf load_b_frag(const __bf16* row, int lane) {
    const int kb = (lane & 16) ? 8 : 0;
    Frag f;
    f.h[0] = *(const v8bf*)(row + kb);
    f.h[1] = *(const v8bf*)(row + kb + 16);
    return f.v;
}

__device__ inline v8f wmma_bf16(v16bf a, v16bf b, v8f c) {
    return __builtin_amdgcn_wmma_f32_16x16x32_bf16(
        false, a, false, b, (short)0, c, false, false);
}

// ---------------------------------------------------------------- TDM (CDNA5)
#if defined(__has_builtin)
#  if __has_builtin(__builtin_amdgcn_tensor_load_to_lds)
#    define HAVE_TDM 1
#  else
#    define HAVE_TDM 0
#  endif
#else
#  define HAVE_TDM 0
#endif

#if __has_include(<hip/amd_detail/amd_gfx1250_TDM.h>)
#  define TDM_6ARG 1   // therock-10.0 headers -> clang-23 6-arg builtin
#else
#  define TDM_6ARG 0   // ROCm 7.2 clang-22 5-arg builtin
#endif

#if HAVE_TDM
#  pragma message("ATHENA-PROBE: HAVE_TDM=1 (tensor_load_to_lds path compiled)")
#else
#  pragma message("ATHENA-PROBE: HAVE_TDM=0 (cooperative-load fallback)")
#endif
#if TDM_6ARG
#  pragma message("ATHENA-PROBE: TDM_6ARG=1")
#else
#  pragma message("ATHENA-PROBE: TDM_6ARG=0")
#endif

#if HAVE_TDM
// D# group1 for a 2D bf16 tile (data_size=1 -> 2 bytes), optional LDS pad.
__device__ inline i32x8 tdm_g1(int tensor_d0, int tensor_d1, int tile_d0,
                               int tile_d1, long long stride0,
                               int pad_en, int pad_interval_c, int pad_amount_c) {
    i32x8 g;
    g[0] = (1 << 16) | (pad_en << 20) | (pad_interval_c << 22) | (pad_amount_c << 25);
    g[1] = (tensor_d0 & 0xFFFF) << 16;                       // tensor_dim0[15:0]
    g[2] = ((tensor_d0 >> 16) & 0xFFFF) | ((tensor_d1 & 0xFFFF) << 16);
    g[3] = ((tensor_d1 >> 16) & 0xFFFF) | (tile_d0 << 16);
    g[4] = (tile_d1 & 0xFFFF);                               // tile_dim1, tile_dim2=0
    g[5] = (int)(stride0 & 0xFFFFFFFFll);                    // tensor_dim0_stride lo
    g[6] = (int)((stride0 >> 32) & 0xFFFF);                  // stride hi, dim1_stride=0
    g[7] = 0;
    return g;
}

__device__ inline void tdm_load(unsigned lds_addr, const void* gptr, i32x8 g1) {
    const unsigned long long ga = (unsigned long long)(uintptr_t)gptr;
    u32x4 g0;
    g0[0] = 1u;                                              // count=1, user desc
    g0[1] = lds_addr;                                        // LDS byte address
    g0[2] = (unsigned)ga;                                    // global_addr[31:0]
    g0[3] = (unsigned)((ga >> 32) & 0x01FFFFFFu) | (2u << 30); // addr[56:32], type=2
    i32x4 z4 = {0, 0, 0, 0};
#if TDM_6ARG
    i32x8 z8 = {0, 0, 0, 0, 0, 0, 0, 0};
    __builtin_amdgcn_tensor_load_to_lds(g0, g1, z4, z4, z8, 0);
#else
    __builtin_amdgcn_tensor_load_to_lds(g0, g1, z4, z4, 0);
#endif
}
#define NBUF 2
#else
#define NBUF 1
#endif // HAVE_TDM

__device__ inline unsigned lds_off(const void* p) {
    return (unsigned)(uintptr_t)p;   // generic LDS pointer: low 32 bits = LDS offset
}

// ---------------------------------------------------------------- conversions
__global__ void cvt_f32_bf16(const float* __restrict__ in, __bf16* __restrict__ out, int n) {
    int i = blockIdx.x * blockDim.x + threadIdx.x;
    if (i < n) out[i] = (__bf16)in[i];
}

// in: [rows(K)][cols(N)] f32  ->  out: [cols(N)][rows(K)] bf16 (transposed)
__global__ void cvtT_f32_bf16(const float* __restrict__ in, __bf16* __restrict__ out,
                              int rows, int cols) {
    int i = blockIdx.x * blockDim.x + threadIdx.x;
    if (i < rows * cols) {
        int k = i / cols, n = i % cols;
        out[(size_t)n * rows + k] = (__bf16)in[i];
    }
}

// ---------------------------------------------------------------- NT GEMM
// C[m][n] = (sum_k A[m][k]*Bt[n][k] + bias[n]) * oscale;  M=4096, N=1024, K=1024.
// Block tile 256x128 (BK=32), 8 waves in 4x2, wave tile 64x64 (16 WMMA/K-step).
// TDM double-buffered: prefetch next tile while computing current.
// MODE 0: bf16 into [B,H,S,64] head layout;  MODE 1: f32 row-major [M][1024].
#define GEMM_K 1024
template <int MODE>
__global__ __launch_bounds__(256)
void gemm_bf16_nt(const __bf16* __restrict__ A, const __bf16* __restrict__ Bt,
                  const float* __restrict__ bias, void* __restrict__ out,
                  float oscale) {
    __shared__ __attribute__((aligned(16))) __bf16 As[NBUF][256][40];
    __shared__ __attribute__((aligned(16))) __bf16 Bs[NBUF][128][40];

    const int tid  = threadIdx.x;
    const int lane = tid & 31;
    const int w    = tid >> 5;          // 8 waves
    const int wm64 = (w >> 1) * 64;     // 0/64/128/192
    const int wn64 = (w & 1) * 64;      // 0/64
    const int bm   = blockIdx.y * 256;
    const int bn   = blockIdx.x * 128;
    const int lrow = lane & 15;

    v8f acc[4][4] = {};

#if HAVE_TDM
    // TDM: pad each 32-elem (64B = 16 DW) row with 4 DW -> 40-elem LDS pitch.
    const i32x8 g1A = tdm_g1(GEMM_K, 4096, 32, 256, GEMM_K, 1, 3, 3);
    const i32x8 g1B = tdm_g1(GEMM_K, 1024, 32, 128, GEMM_K, 1, 3, 3);
    const bool issuer = (w == 0);
    if (issuer) {
        tdm_load(lds_off(&As[0][0][0]), A + (size_t)bm * GEMM_K, g1A);
        tdm_load(lds_off(&Bs[0][0][0]), Bt + (size_t)bn * GEMM_K, g1B);
    }
#else
    const int brow = tid >> 1;          // B tile: 128 rows, half row each
    const int bcol = (tid & 1) * 16;
#endif

    for (int it = 0; it < GEMM_K / 32; ++it) {
        const int k0 = it * 32;
        const int p  = it & (NBUF - 1);
        __syncthreads();                 // all waves done with buffer we refill
#if HAVE_TDM
        if (issuer) {
            if (it + 1 < GEMM_K / 32) {  // prefetch next tile into other buffer
                tdm_load(lds_off(&As[1 - p][0][0]),
                         A + (size_t)bm * GEMM_K + k0 + 32, g1A);
                tdm_load(lds_off(&Bs[1 - p][0][0]),
                         Bt + (size_t)bn * GEMM_K + k0 + 32, g1B);
                __builtin_amdgcn_s_wait_tensorcnt(2);  // current 2 ops complete
            } else {
                __builtin_amdgcn_s_wait_tensorcnt(0);
            }
        }
        __syncthreads();
#else
        {   // cooperative fallback: A row per thread, B half-row per thread
            const __bf16* ag = A + (size_t)(bm + tid) * GEMM_K + k0;
            #pragma unroll
            for (int c = 0; c < 32; c += 8)
                *(v8bf*)&As[0][tid][c] = *(const v8bf*)(ag + c);
            const __bf16* bg = Bt + (size_t)(bn + brow) * GEMM_K + k0 + bcol;
            *(v8bf*)&Bs[0][brow][bcol]     = *(const v8bf*)(bg);
            *(v8bf*)&Bs[0][brow][bcol + 8] = *(const v8bf*)(bg + 8);
        }
        __syncthreads();
#endif

        v16bf afr[4], bfr[4];
        #pragma unroll
        for (int mi = 0; mi < 4; ++mi)
            afr[mi] = load_a_frag(&As[p][wm64 + mi * 16 + lrow][0], lane);
        #pragma unroll
        for (int ni = 0; ni < 4; ++ni)
            bfr[ni] = load_b_frag(&Bs[p][wn64 + ni * 16 + lrow][0], lane);

        #pragma unroll
        for (int mi = 0; mi < 4; ++mi)
            #pragma unroll
            for (int ni = 0; ni < 4; ++ni)
                acc[mi][ni] = wmma_bf16(afr[mi], bfr[ni], acc[mi][ni]);
    }

    // epilogue: C tile layout -> VGPR e = row (e + 8*(lane>=16)), col = lane%16
    const int half = lane >> 4;
    const int col  = lane & 15;
    #pragma unroll
    for (int ni = 0; ni < 4; ++ni) {
        const int gn = bn + wn64 + ni * 16 + col;
        const float bv = bias[gn];
        #pragma unroll
        for (int mi = 0; mi < 4; ++mi) {
            #pragma unroll
            for (int e = 0; e < 8; ++e) {
                const int gm = bm + wm64 + mi * 16 + e + half * 8;
                const float v = (acc[mi][ni][e] + bv) * oscale;
                if (MODE == 0) {
                    __bf16* o = (__bf16*)out;
                    const int b = gm >> 11, s = gm & 2047;
                    const int h = gn >> 6,  d = gn & 63;
                    o[(((size_t)(b * 16 + h)) * 2048 + s) * 64 + d] = (__bf16)v;
                } else {
                    float* o = (float*)out;
                    o[(size_t)gm * 1024 + gn] = v;
                }
            }
        }
    }
}

// ---------------------------------------------------------------- attention
// One block = 128 query rows of one (b,h); 8 waves x 16 rows.
// q is pre-scaled by 1/sqrt(64).  Max-free streaming softmax (f32 accum,
// scores have tiny dynamic range): P = exp(s); row-sums accumulated by an
// extra WMMA against an all-ones B fragment (no cross-lane shuffles at all).
__global__ __launch_bounds__(256)
void attn_flash(const __bf16* __restrict__ qh, const __bf16* __restrict__ kh,
                const __bf16* __restrict__ vh, __bf16* __restrict__ ctx) {
    __shared__ __attribute__((aligned(16))) __bf16 Ks[32][72];
    __shared__ __attribute__((aligned(16))) __bf16 Vt[64][40];
    __shared__ __attribute__((aligned(16))) __bf16 Pw[8][16][40];

    const int tid  = threadIdx.x;
    const int lane = tid & 31;
    const int w    = tid >> 5;
    const int qt   = blockIdx.x;
    const int h    = blockIdx.y;
    const int b    = blockIdx.z;
    const int lrow = lane & 15;
    const int half = lane >> 4;

    const size_t head = ((size_t)(b * 16 + h)) * 2048 * 64;
    const __bf16* qb = qh + head;
    const __bf16* kb = kh + head;
    const __bf16* vb = vh + head;

    v16bf qa0, qa1;
    {
        const __bf16* qrow = qb + (size_t)(qt * 128 + w * 16 + lrow) * 64;
        qa0 = load_a_frag(qrow, lane);
        qa1 = load_a_frag(qrow + 32, lane);
    }

    Frag onef;                       // all-ones 32x16 B fragment (row-sum WMMA)
    #pragma unroll
    for (int i = 0; i < 16; ++i) onef.v[i] = (__bf16)1.0f;

    v8f o[4] = {};
    v8f osum = {};                   // every column holds the running row-sum

    const int ldr = tid >> 3;        // key row 0..31
    const int ldc = (tid & 7) * 8;   // d offset 0..56

    for (int kt = 0; kt < 2048; kt += 32) {
        __syncthreads();
        {
            const v8bf kv = *(const v8bf*)(kb + (size_t)(kt + ldr) * 64 + ldc);
            *(v8bf*)&Ks[ldr][ldc] = kv;
            const v8bf vv = *(const v8bf*)(vb + (size_t)(kt + ldr) * 64 + ldc);
            #pragma unroll
            for (int i = 0; i < 8; ++i) Vt[ldc + i][ldr] = vv[i];
        }
        __syncthreads();

        // scores (q already carries 1/sqrt(dk)); two 16x16 tiles, K=64 chained
        v8f s[2];
        #pragma unroll
        for (int j = 0; j < 2; ++j) {
            v16bf bk0 = load_b_frag(&Ks[j * 16 + lrow][0], lane);
            v16bf bk1 = load_b_frag(&Ks[j * 16 + lrow][32], lane);
            v8f c = {};
            c = wmma_bf16(qa0, bk0, c);
            c = wmma_bf16(qa1, bk1, c);
            s[j] = c;
        }

        // P = exp(s) -> LDS (C layout) -> reload as A fragment
        #pragma unroll
        for (int e = 0; e < 8; ++e) {
            Pw[w][e + half * 8][lrow]      = (__bf16)__expf(s[0][e]);
            Pw[w][e + half * 8][16 + lrow] = (__bf16)__expf(s[1][e]);
        }
        asm volatile("s_wait_dscnt 0" ::: "memory");
        v16bf pf = load_a_frag(&Pw[w][lrow][0], lane);

        // O += P x V (4 d-groups) ; row-sum += P x ones
        #pragma unroll
        for (int t = 0; t < 4; ++t) {
            v16bf vf = load_b_frag(&Vt[t * 16 + lrow][0], lane);
            o[t] = wmma_bf16(pf, vf, o[t]);
        }
        osum = wmma_bf16(pf, onef.v, osum);
    }

    const int gmb = b * 2048 + qt * 128 + w * 16;
    #pragma unroll
    for (int e = 0; e < 8; ++e) {
        const float inv = 1.0f / osum[e];
        const int row = gmb + e + half * 8;
        #pragma unroll
        for (int t = 0; t < 4; ++t) {
            const int cc = h * 64 + t * 16 + lrow;
            ctx[(size_t)row * 1024 + cc] = (__bf16)(o[t][e] * inv);
        }
    }
}

// ---------------------------------------------------------------- launch
extern "C" void kernel_launch(void* const* d_in, const int* in_sizes, int n_in,
                              void* d_out, int out_size, void* d_ws, size_t ws_size,
                              hipStream_t stream) {
    (void)in_sizes; (void)n_in; (void)out_size; (void)ws_size;
    const float* Q  = (const float*)d_in[0];
    const float* K  = (const float*)d_in[1];
    const float* V  = (const float*)d_in[2];
    const float* WQ = (const float*)d_in[3];
    const float* bQ = (const float*)d_in[4];
    const float* WK = (const float*)d_in[5];
    const float* bK = (const float*)d_in[6];
    const float* WV = (const float*)d_in[7];
    const float* bV = (const float*)d_in[8];
    const float* WO = (const float*)d_in[9];
    const float* bO = (const float*)d_in[10];

    const size_t MB = 1u << 20;
    char* ws = (char*)d_ws;
    __bf16* Xq  = (__bf16*)(ws + 0 * MB);
    __bf16* Xk  = (__bf16*)(ws + 8 * MB);
    __bf16* Xv  = (__bf16*)(ws + 16 * MB);
    __bf16* Wqt = (__bf16*)(ws + 24 * MB);
    __bf16* Wkt = (__bf16*)(ws + 26 * MB);
    __bf16* Wvt = (__bf16*)(ws + 28 * MB);
    __bf16* Wot = (__bf16*)(ws + 30 * MB);
    __bf16* qhp = (__bf16*)(ws + 32 * MB);
    __bf16* khp = (__bf16*)(ws + 40 * MB);
    __bf16* vhp = (__bf16*)(ws + 48 * MB);
    __bf16* ctx = (__bf16*)(ws + 56 * MB);

    const int nX = 4096 * 1024;
    const int nW = 1024 * 1024;
    cvt_f32_bf16 <<<(nX + 255) / 256, 256, 0, stream>>>(Q, Xq, nX);
    cvt_f32_bf16 <<<(nX + 255) / 256, 256, 0, stream>>>(K, Xk, nX);
    cvt_f32_bf16 <<<(nX + 255) / 256, 256, 0, stream>>>(V, Xv, nX);
    cvtT_f32_bf16<<<(nW + 255) / 256, 256, 0, stream>>>(WQ, Wqt, 1024, 1024);
    cvtT_f32_bf16<<<(nW + 255) / 256, 256, 0, stream>>>(WK, Wkt, 1024, 1024);
    cvtT_f32_bf16<<<(nW + 255) / 256, 256, 0, stream>>>(WV, Wvt, 1024, 1024);
    cvtT_f32_bf16<<<(nW + 255) / 256, 256, 0, stream>>>(WO, Wot, 1024, 1024);

    dim3 gg(8, 16);   // N/128, M/256
    gemm_bf16_nt<0><<<gg, 256, 0, stream>>>(Xq, Wqt, bQ, qhp, 0.125f); // q * 1/sqrt(dk)
    gemm_bf16_nt<0><<<gg, 256, 0, stream>>>(Xk, Wkt, bK, khp, 1.0f);
    gemm_bf16_nt<0><<<gg, 256, 0, stream>>>(Xv, Wvt, bV, vhp, 1.0f);

    dim3 ga(16, 16, 2);   // S/128 q-tiles, H, B
    attn_flash<<<ga, 256, 0, stream>>>(qhp, khp, vhp, ctx);

    gemm_bf16_nt<1><<<gg, 256, 0, stream>>>(ctx, Wot, bO, d_out, 1.0f);
}
